// LSTM_49916109914652
// MI455X (gfx1250) — matrix operand order repaired and checked by reading, not compiled, not run on hardware
//
#include <hip/hip_runtime.h>
#include <hip/hip_bf16.h>
#include <math.h>

// ---------------------------------------------------------------------------
// Problem constants (from reference)
// ---------------------------------------------------------------------------
#define CB   1024        // B
#define CP   16          // P
#define CL   8           // L
#define CH   60          // H
#define CG   240         // 4*H
#define CED  300         // word emb dim
#define CIN  310         // ED+POSD+DEPD+DIRD
#define CINP 320         // padded K for layer0 input GEMM
#define CHP  64          // padded K for H-sized GEMMs
#define CNREL 5
#define NSEQ (CB * CP)           // 16384
#define NL   (NSEQ * CL)         // 131072
#define MAXK CINP                // largest K staged in LDS (16 x 320 f16 = 10 KB)

typedef __attribute__((ext_vector_type(16))) _Float16 v16h;
typedef __attribute__((ext_vector_type(8)))  _Float16 v8h;
typedef __attribute__((ext_vector_type(8)))  float    v8f;
typedef int vasync_t __attribute__((vector_size(4 * sizeof(int))));   // matches builtin param

#define AS1 __attribute__((address_space(1)))
#define AS3 __attribute__((address_space(3)))

// Async global->LDS staging of 16 bytes, with fallback (plain load + ds_store).
__device__ __forceinline__ void stage16B(const _Float16* src, _Float16* lds)
{
#if __has_builtin(__builtin_amdgcn_global_load_async_to_lds_b128)
    _Float16* s = const_cast<_Float16*>(src);
    __builtin_amdgcn_global_load_async_to_lds_b128(
        (AS1 vasync_t*)s, (AS3 vasync_t*)lds, 0, 0);
#else
    *(v8h*)lds = *(const v8h*)src;
#endif
}

__device__ __forceinline__ void wait_async_copies()
{
#if __has_builtin(__builtin_amdgcn_global_load_async_to_lds_b128)
#if __has_builtin(__builtin_amdgcn_s_wait_asynccnt)
    __builtin_amdgcn_s_wait_asynccnt(0);
#else
    asm volatile("s_wait_asynccnt 0" ::: "memory");
#endif
#endif
}

// ---------------------------------------------------------------------------
// WMMA GEMM with LDS-staged weights:
//   Cout[m,n] = (optional Cin[m,n]) + sum_k A[m,k] * Bw[n,k]  (+ bias[n])
//   A  : f16, logical row m at A + m*lda (K padded to mult of 32)
//   Bw : f16, N x K row-major, rows CONTIGUOUS (ldb == K)
//   C  : f32, logical row m at C + m*ldc (strided timestep slots supported)
// Block = 128 threads (4 waves) computes a 256-row x 16-col super-tile:
//   - whole block async-copies the 16 x K weight slab into LDS once
//   - each wave computes four 16x16 sub-tiles, reusing the LDS B fragment
// Grid is sized exactly, EXEC stays all-ones around every v_wmma.
// ---------------------------------------------------------------------------
__global__ void wmma_gemm_f16f32_kernel(
    const _Float16* __restrict__ A,  long lda,
    const _Float16* __restrict__ Bw,     // ldb == K
    const float*    __restrict__ bias,   // length N or nullptr
    const float*    __restrict__ Cin,    // nullptr or same layout as Cout
    float*          __restrict__ Cout,   long ldc,
    int M, int N, int K)
{
    __shared__ _Float16 Bs[16 * MAXK];

    const int ntiles = N >> 4;
    const int blk = blockIdx.x;
    const int ms  = blk / ntiles;        // 256-row super-tile index
    const int nt  = blk - ms * ntiles;
    const int n0  = nt << 4;

    // ---- cooperative async stage of the 16 x K weight slab ----
    {
        const _Float16* src = Bw + (size_t)n0 * K;   // contiguous 16*K halves
        const int chunks = (16 * K) >> 3;            // 16-byte chunks
        for (int i = threadIdx.x; i < chunks; i += blockDim.x)
            stage16B(src + i * 8, Bs + i * 8);
        wait_async_copies();
        __syncthreads();
    }

    const int wv    = threadIdx.x >> 5;              // wave 0..3
    const int lane  = threadIdx.x & 31;
    const int lhalf = lane >> 4;                     // 0: lanes 0-15, 1: 16-31
    const int lrow  = lane & 15;
    const int m0    = (ms << 8) + (wv << 6);         // this wave's 64-row strip

    // A fragment base (ISA 16x32 f16 layout): lane<16 K={0..7,16..23},
    // lane>=16 K={8..15,24..31} of row m0+lrow -> two b128 loads per sub-tile.
    const _Float16* aptr = A + (size_t)(m0 + lrow) * lda + (lhalf ? 8 : 0);
    // B fragment from LDS: lane holds column n0+lrow, K={0..7,16..23} (lane<16)
    // or K={8..15,24..31} (lane>=16) -> two ds_load_b128.
    const _Float16* bls = Bs + (size_t)lrow * K + (lhalf ? 8 : 0);

    v8f c[4] = {};
    for (int k0 = 0; k0 < K; k0 += 32) {
        if (k0 + 32 < K)
            __builtin_prefetch(aptr + k0 + 32, 0, 3);   // global_prefetch_b8 (near)
        v8h blo = *(const v8h*)(bls + k0);
        v8h bhi = *(const v8h*)(bls + k0 + 16);
        v16h b;
#pragma unroll
        for (int i = 0; i < 8; ++i) { b[i] = blo[i]; b[8 + i] = bhi[i]; }
#pragma unroll
        for (int s = 0; s < 4; ++s) {                   // 4 M sub-tiles reuse B
            const _Float16* ap = aptr + (size_t)(s * 16) * lda + k0;
            v8h alo = *(const v8h*)(ap);
            v8h ahi = *(const v8h*)(ap + 16);
            v16h a;
#pragma unroll
            for (int i = 0; i < 8; ++i) { a[i] = alo[i]; a[8 + i] = ahi[i]; }
            c[s] = __builtin_amdgcn_wmma_f32_16x16x32_f16(
                       false, a, false, b, (short)0, c[s], false, false);
        }
    }

    const float bv = bias ? bias[n0 + lrow] : 0.0f;
#pragma unroll
    for (int s = 0; s < 4; ++s) {
        const int rbase = m0 + s * 16 + (lhalf << 3);   // D rows per ISA layout
#pragma unroll
        for (int r = 0; r < 8; ++r) {
            size_t off = (size_t)(rbase + r) * ldc + (size_t)(n0 + lrow);
            float v = c[s][r] + bv;
            if (Cin) v += Cin[off];
            Cout[off] = v;
        }
    }
}

// ---------------------------------------------------------------------------
// Embedding gather + concat -> padded f16 feature rows (NL x 320)
// ---------------------------------------------------------------------------
__global__ void embed_gather_kernel(
    const int* __restrict__ wi, const int* __restrict__ pi,
    const int* __restrict__ di, const int* __restrict__ ri,
    const float* __restrict__ we, const float* __restrict__ pe,
    const float* __restrict__ de, const float* __restrict__ re,
    _Float16* __restrict__ X)
{
    long tid = (long)blockIdx.x * blockDim.x + threadIdx.x;
    const long total = (long)NL * 20;             // 20 segments of 16 halves per row
    if (tid >= total) return;
    int  seg = (int)(tid % 20);
    long row = tid / 20;
    int w = wi[row], p = pi[row], d = di[row], r = ri[row];
    _Float16* dst = X + row * CINP + seg * 16;
#pragma unroll
    for (int i = 0; i < 16; ++i) {
        int k = seg * 16 + i;
        float v;
        if      (k < 300) v = we[(size_t)w * CED + k];
        else if (k < 304) v = pe[p * 4 + (k - 300)];
        else if (k < 309) v = de[d * 5 + (k - 304)];
        else if (k < 310) v = re[r];
        else              v = 0.0f;
        dst[i] = (_Float16)v;
    }
}

// Convert f32 (N x K) weights to f16 (N x Kp) with zero K-padding.
__global__ void convert_pad_kernel(const float* __restrict__ src,
                                   _Float16* __restrict__ dst,
                                   int N, int K, int Kp)
{
    int tid = blockIdx.x * blockDim.x + threadIdx.x;
    if (tid >= N * Kp) return;
    int k  = tid % Kp;
    int nn = tid / Kp;
    dst[tid] = (_Float16)((k < K) ? src[(size_t)nn * K + k] : 0.0f);
}

__global__ void bias_sum_kernel(const float* __restrict__ a,
                                const float* __restrict__ b,
                                float* __restrict__ o, int n)
{
    int tid = blockIdx.x * blockDim.x + threadIdx.x;
    if (tid < n) o[tid] = a[tid] + b[tid];
}

__global__ void zero_f32_kernel(float* __restrict__ p, long n)
{
    long tid = (long)blockIdx.x * blockDim.x + threadIdx.x;
    if (tid < n) p[tid] = 0.0f;
}

// ---------------------------------------------------------------------------
// LSTM pointwise gates for one timestep.
// ---------------------------------------------------------------------------
__global__ void lstm_gate_kernel(const float* __restrict__ G, long grs,
                                 float* __restrict__ c,
                                 _Float16* __restrict__ hf16, long hrs,
                                 float* __restrict__ hf32, long h32rs)
{
    int tid = blockIdx.x * blockDim.x + threadIdx.x;
    if (tid >= NSEQ * CH) return;
    int j = tid % CH;
    int s = tid / CH;
    const float* g = G + (size_t)s * grs;
    float gi = g[j], gf = g[CH + j], gg = g[2 * CH + j], go = g[3 * CH + j];
    float si = 1.0f / (1.0f + expf(-gi));
    float sf = 1.0f / (1.0f + expf(-gf));
    float so = 1.0f / (1.0f + expf(-go));
    float cs = sf * c[tid] + si * tanhf(gg);
    float h  = so * tanhf(cs);
    c[tid] = cs;
    hf16[(size_t)s * hrs + j] = (_Float16)h;
    if (j < CHP - CH) hf16[(size_t)s * hrs + CH + j] = (_Float16)0.0f;  // keep K-pad zero
    if (hf32) hf32[(size_t)s * h32rs + j] = h;
}

// last-timestep select + counts-weighted mean over P -> path_emb (B x 60)
__global__ void select_avg_kernel(const float* __restrict__ h2,
                                  const int*   __restrict__ lens,
                                  const float* __restrict__ counts,
                                  float* __restrict__ path)
{
    int tid = blockIdx.x * blockDim.x + threadIdx.x;
    if (tid >= CB * CH) return;
    int j = tid % CH;
    int b = tid / CH;
    float num = 0.0f, den = 0.0f;
    for (int p = 0; p < CP; ++p) {
        int s = b * CP + p;
        int t = lens[s] - 1;
        float cw = counts[s];
        num += cw * h2[((size_t)s * CL + t) * CH + j];
        den += cw;
    }
    path[tid] = num / den;
}

// feat = [word_emb[pair0] | path | word_emb[pair1]] ; logits = feat@W.T + b ; softmax
__global__ void final_softmax_kernel(const int* __restrict__ pair,
                                     const float* __restrict__ we,
                                     const float* __restrict__ path,
                                     const float* __restrict__ W,
                                     const float* __restrict__ bv,
                                     float* __restrict__ out)
{
    int b = blockIdx.x * blockDim.x + threadIdx.x;
    if (b >= CB) return;
    const float* xw = we + (size_t)pair[b * 2 + 0] * CED;
    const float* yw = we + (size_t)pair[b * 2 + 1] * CED;
    const float* pb = path + (size_t)b * CH;
    float lg[CNREL];
#pragma unroll
    for (int r = 0; r < CNREL; ++r) {
        const float* wr = W + (size_t)r * (CED + CH + CED);
        float acc = bv[r];
        for (int k = 0; k < CED; ++k) acc += xw[k] * wr[k];
        for (int k = 0; k < CH;  ++k) acc += pb[k] * wr[CED + k];
        for (int k = 0; k < CED; ++k) acc += yw[k] * wr[CED + CH + k];
        lg[r] = acc;
    }
    float mx = lg[0];
#pragma unroll
    for (int r = 1; r < CNREL; ++r) mx = fmaxf(mx, lg[r]);
    float sum = 0.0f;
#pragma unroll
    for (int r = 0; r < CNREL; ++r) { lg[r] = expf(lg[r] - mx); sum += lg[r]; }
    float inv = 1.0f / sum;
#pragma unroll
    for (int r = 0; r < CNREL; ++r) out[(size_t)b * CNREL + r] = lg[r] * inv;
}

// ---------------------------------------------------------------------------
// Host-side orchestration
// ---------------------------------------------------------------------------
extern "C" void kernel_launch(void* const* d_in, const int* in_sizes, int n_in,
                              void* d_out, int out_size, void* d_ws, size_t ws_size,
                              hipStream_t stream)
{
    const int*   word_idx = (const int*)  d_in[0];
    const int*   pos_idx  = (const int*)  d_in[1];
    const int*   dep_idx  = (const int*)  d_in[2];
    const int*   dir_idx  = (const int*)  d_in[3];
    const int*   lengths  = (const int*)  d_in[4];
    const float* counts   = (const float*)d_in[5];
    const int*   pair_idx = (const int*)  d_in[6];
    const float* word_emb = (const float*)d_in[7];
    const float* pos_emb  = (const float*)d_in[8];
    const float* dep_emb  = (const float*)d_in[9];
    const float* dir_emb  = (const float*)d_in[10];
    const float* Wih0 = (const float*)d_in[11];
    const float* Whh0 = (const float*)d_in[12];
    const float* bih0 = (const float*)d_in[13];
    const float* bhh0 = (const float*)d_in[14];
    const float* Wih1 = (const float*)d_in[15];
    const float* Whh1 = (const float*)d_in[16];
    const float* bih1 = (const float*)d_in[17];
    const float* bhh1 = (const float*)d_in[18];
    const float* Wc   = (const float*)d_in[19];
    const float* bc   = (const float*)d_in[20];

    // ---- workspace carving (256B aligned slabs) ----
    char*  base = (char*)d_ws;
    size_t off  = 0;
    auto alloc = [&](size_t bytes) -> char* {
        char* p = base + off;
        off = (off + bytes + 255) & ~(size_t)255;
        return p;
    };
    _Float16* Xf16   = (_Float16*)alloc((size_t)NL * CINP * 2);   // 80 MB
    _Float16* W0f    = (_Float16*)alloc((size_t)CG * CINP * 2);
    _Float16* Wh0f   = (_Float16*)alloc((size_t)CG * CHP * 2);
    _Float16* W1f    = (_Float16*)alloc((size_t)CG * CHP * 2);
    _Float16* Wh1f   = (_Float16*)alloc((size_t)CG * CHP * 2);
    float*    bias0  = (float*)   alloc((size_t)CG * 4);
    float*    bias1  = (float*)   alloc((size_t)CG * 4);
    float*    Gbuf   = (float*)   alloc((size_t)NL * CG * 4);     // 126 MB (reused L0/L1)
    _Float16* h1all  = (_Float16*)alloc((size_t)NL * CHP * 2);    // 16.8 MB
    _Float16* h2f16  = (_Float16*)alloc((size_t)NL * CHP * 2);
    float*    h2f32  = (float*)   alloc((size_t)NL * CH * 4);     // 31.5 MB
    float*    cell0  = (float*)   alloc((size_t)NSEQ * CH * 4);
    float*    cell1  = (float*)   alloc((size_t)NSEQ * CH * 4);
    float*    path   = (float*)   alloc((size_t)CB * CH * 4);
    (void)ws_size; (void)n_in; (void)in_sizes; (void)out_size;

    const int T = 256;
    auto blocks = [](long n, int t) { return (int)((n + t - 1) / t); };

    // ---- weight conversion / bias prep ----
    convert_pad_kernel<<<blocks((long)CG * CINP, T), T, 0, stream>>>(Wih0, W0f,  CG, CIN, CINP);
    convert_pad_kernel<<<blocks((long)CG * CHP,  T), T, 0, stream>>>(Whh0, Wh0f, CG, CH,  CHP);
    convert_pad_kernel<<<blocks((long)CG * CHP,  T), T, 0, stream>>>(Wih1, W1f,  CG, CH,  CHP);
    convert_pad_kernel<<<blocks((long)CG * CHP,  T), T, 0, stream>>>(Whh1, Wh1f, CG, CH,  CHP);
    bias_sum_kernel<<<1, CG, 0, stream>>>(bih0, bhh0, bias0, CG);
    bias_sum_kernel<<<1, CG, 0, stream>>>(bih1, bhh1, bias1, CG);
    zero_f32_kernel<<<blocks((long)NSEQ * CH, T), T, 0, stream>>>(cell0, (long)NSEQ * CH);
    zero_f32_kernel<<<blocks((long)NSEQ * CH, T), T, 0, stream>>>(cell1, (long)NSEQ * CH);

    // ---- embedding gather -> f16 features (all seqs, all timesteps) ----
    embed_gather_kernel<<<blocks((long)NL * 20, T), T, 0, stream>>>(
        word_idx, pos_idx, dep_idx, dir_idx,
        word_emb, pos_emb, dep_emb, dir_emb, Xf16);

    const int BT = 128;                                // 4 waves per block
    const int bigBlocks = (NL   / 256) * (CG / 16);    // 512 * 15
    const int recBlocks = (NSEQ / 256) * (CG / 16);    //  64 * 15

    // ---- layer 0: batched input projection over all 8 timesteps ----
    wmma_gemm_f16f32_kernel<<<bigBlocks, BT, 0, stream>>>(
        Xf16, CINP, W0f, bias0, nullptr, Gbuf, CG, NL, CG, CINP);

    // ---- layer 0: serial scan (only h@Whh.T is sequential) ----
    for (int t = 0; t < CL; ++t) {
        if (t > 0) {
            wmma_gemm_f16f32_kernel<<<recBlocks, BT, 0, stream>>>(
                h1all + (size_t)(t - 1) * CHP, (long)CL * CHP,
                Wh0f, nullptr,
                Gbuf + (size_t)t * CG, Gbuf + (size_t)t * CG, (long)CL * CG,
                NSEQ, CG, CHP);
        }
        lstm_gate_kernel<<<blocks((long)NSEQ * CH, T), T, 0, stream>>>(
            Gbuf + (size_t)t * CG, (long)CL * CG, cell0,
            h1all + (size_t)t * CHP, (long)CL * CHP, nullptr, 0);
    }

    // ---- layer 1: batched input projection over all timesteps ----
    wmma_gemm_f16f32_kernel<<<bigBlocks, BT, 0, stream>>>(
        h1all, CHP, W1f, bias1, nullptr, Gbuf, CG, NL, CG, CHP);

    // ---- layer 1: serial scan ----
    for (int t = 0; t < CL; ++t) {
        if (t > 0) {
            wmma_gemm_f16f32_kernel<<<recBlocks, BT, 0, stream>>>(
                h2f16 + (size_t)(t - 1) * CHP, (long)CL * CHP,
                Wh1f, nullptr,
                Gbuf + (size_t)t * CG, Gbuf + (size_t)t * CG, (long)CL * CG,
                NSEQ, CG, CHP);
        }
        lstm_gate_kernel<<<blocks((long)NSEQ * CH, T), T, 0, stream>>>(
            Gbuf + (size_t)t * CG, (long)CL * CG, cell1,
            h2f16 + (size_t)t * CHP, (long)CL * CHP,
            h2f32 + (size_t)t * CH, (long)CL * CH);
    }

    // ---- tail: select last step, mean over P, classifier + softmax ----
    select_avg_kernel<<<blocks((long)CB * CH, T), T, 0, stream>>>(
        h2f32, lengths, counts, path);
    final_softmax_kernel<<<blocks(CB, T), T, 0, stream>>>(
        pair_idx, word_emb, path, Wc, bc, (float*)d_out);
}
